// CategoryRouter_34668976013723
// MI455X (gfx1250) — matrix operand order repaired and checked by reading, not compiled, not run on hardware
//
#include <hip/hip_runtime.h>

// ---------------------------------------------------------------------------
// CategoryRouter, CDNA5 (gfx1250): routed 2-layer MLP head + softmax.
//   out[t] = softmax( relu(x[t] @ W1[r] + b1[r]) @ W2[r] + b2[r] ), r = route[t]
// Pipeline:
//   1) convert X -> bf16 ; transpose-convert W1,W2 -> bf16 K-major (workspace)
//   2) bin tokens by route ; per-route padded row bases (device prefix)
//   3) GEMM1: H = relu(Xb @ W1t + b1)  (bf16 in ws, binned row order)
//   4) GEMM2: logits = H @ W2t + b2 -> d_out (fp32)
//   5) softmax in place over V per token
// GEMM inner loops: contiguous 16B fragment loads + v_wmma_f32_16x16x32_bf16,
// 64x64 WMMA tile per wave (4x4), 128x256 per block.
// ---------------------------------------------------------------------------

#define R_    8
#define D_    768
#define F_    3072
#define V_    2048
#define NTOK  4096
#define MAXPR 4096
#define HROWS 5120   // 4096 tokens + 8 routes * 127 max padding, rounded

typedef __attribute__((ext_vector_type(16))) __bf16 v16bf;
typedef __attribute__((ext_vector_type(8)))  float  v8f;

__device__ __forceinline__ unsigned short f2bf(float f) {
  unsigned int u = __float_as_uint(f);
  u += 0x7FFFu + ((u >> 16) & 1u);
  return (unsigned short)(u >> 16);
}

union FragBF16 { unsigned short us[16]; uint4 q[2]; v16bf v; };

// A fragment (16x32 bf16) from row-major bf16, per-lane row base.
// Lane l: m=l&15, h=l>>4. e=0..7 -> K=k0+8h+e ; e=8..15 -> K=k0+16+8h+(e-8).
__device__ __forceinline__ v16bf load_a(const unsigned short* __restrict__ rowp,
                                        int k0, int h) {
  FragBF16 a;
  const unsigned short* p = rowp + k0 + 8 * h;
  a.q[0] = *(const uint4*)(p);
  a.q[1] = *(const uint4*)(p + 16);
  return a.v;
}

// B fragment (32x16 bf16) from K-major (transposed) weights, per-lane col base.
// Lane l: n=l&15, h=l>>4. e=0..7 -> K=k0+8h+e ; e=8..15 -> K=k0+16+8h+(e-8).
__device__ __forceinline__ v16bf load_b(const unsigned short* __restrict__ colp,
                                        int k0, int h) {
  FragBF16 b;
  const unsigned short* p = colp + k0 + 8 * h;
  b.q[0] = *(const uint4*)(p);
  b.q[1] = *(const uint4*)(p + 16);
  return b.v;
}

// ---------------------------------------------------------------------------
// Elementwise fp32 -> bf16 convert (for X).
// ---------------------------------------------------------------------------
__global__ __launch_bounds__(256) void convert_kernel(
    const float* __restrict__ src, unsigned short* __restrict__ dst, int n) {
  int i = blockIdx.x * 256 + threadIdx.x;
  if (i < n) dst[i] = f2bf(src[i]);
}

// ---------------------------------------------------------------------------
// Tiled transpose + convert: src fp32 [Z][K][N] -> dst bf16 [Z][N][K].
// ---------------------------------------------------------------------------
__global__ __launch_bounds__(256) void transpose_convert_kernel(
    const float* __restrict__ src, unsigned short* __restrict__ dst,
    int K, int N) {
  __shared__ float tile[32][33];
  const float* s = src + (size_t)blockIdx.z * K * N;
  unsigned short* d = dst + (size_t)blockIdx.z * N * K;
  const int n0 = blockIdx.x * 32, k0 = blockIdx.y * 32;
  const int tx = threadIdx.x, ty = threadIdx.y;
#pragma unroll
  for (int j = 0; j < 32; j += 8)
    tile[ty + j][tx] = s[(size_t)(k0 + ty + j) * N + (n0 + tx)];
  __syncthreads();
#pragma unroll
  for (int j = 0; j < 32; j += 8)
    d[(size_t)(n0 + ty + j) * K + (k0 + tx)] = f2bf(tile[tx][ty + j]);
}

// ---------------------------------------------------------------------------
// Bin tokens by route; tiny prefix kernel for 128-padded row bases.
// ---------------------------------------------------------------------------
__global__ __launch_bounds__(256) void route_bin_kernel(
    const int* __restrict__ rid, int* __restrict__ cnt, int* __restrict__ perm) {
  int t = blockIdx.x * 256 + threadIdx.x;
  if (t >= NTOK) return;
  int r = rid[t];
  int slot = atomicAdd(&cnt[r], 1);
  perm[r * MAXPR + slot] = t;
}

__global__ void route_base_kernel(const int* __restrict__ cnt,
                                  int* __restrict__ base) {
  if (threadIdx.x == 0 && blockIdx.x == 0) {
    int acc = 0;
    for (int r = 0; r < R_; ++r) {
      base[r] = acc;
      acc += ((cnt[r] + 127) >> 7) << 7;
    }
  }
}

// ---------------------------------------------------------------------------
// GEMM1: H[binned rows, F] = relu(Xb @ W1t + b1), bf16 out.
// Block = 128 tokens x 256 F cols, 8 waves (2 M-groups x 4 N-groups),
// wave = 64x64 = 4x4 WMMA tiles, K = 768.  grid = (F/256, 32, R)
// ---------------------------------------------------------------------------
__global__ __launch_bounds__(256) void gemm1_kernel(
    const unsigned short* __restrict__ Xb, const unsigned short* __restrict__ W1t,
    const float* __restrict__ b1, const int* __restrict__ cnt,
    const int* __restrict__ base, const int* __restrict__ perm,
    unsigned short* __restrict__ H) {
  const int r = blockIdx.z;
  const int c = cnt[r];
  const int mblk = blockIdx.y * 128;
  if (mblk >= c) return;

  const int tid = threadIdx.x, wave = tid >> 5, lane = tid & 31;
  const int h = lane >> 4, nn = lane & 15;
  const int m0 = mblk + (wave & 1) * 64;
  const int n0 = blockIdx.x * 256 + (wave >> 1) * 64;
  const int* permr = perm + r * MAXPR;

  const unsigned short* aptr[4];
#pragma unroll
  for (int ms = 0; ms < 4; ++ms) {
    int row = m0 + ms * 16 + nn;
    int tok = (row < c) ? permr[row] : 0;   // clamp: finite garbage, only lands in H padding rows
    aptr[ms] = Xb + (size_t)tok * D_;
  }
  const unsigned short* bptr[4];
#pragma unroll
  for (int ns = 0; ns < 4; ++ns)
    bptr[ns] = W1t + ((size_t)r * F_ + (n0 + ns * 16 + nn)) * D_;

  const v8f zero = {};
  v8f acc[4][4];
#pragma unroll
  for (int ms = 0; ms < 4; ++ms)
#pragma unroll
    for (int ns = 0; ns < 4; ++ns) acc[ms][ns] = zero;

  for (int k0 = 0; k0 < D_; k0 += 32) {
    v16bf a[4];
#pragma unroll
    for (int ms = 0; ms < 4; ++ms) a[ms] = load_a(aptr[ms], k0, h);
#pragma unroll
    for (int ns = 0; ns < 4; ++ns) {
      v16bf b = load_b(bptr[ns], k0, h);
#pragma unroll
      for (int ms = 0; ms < 4; ++ms)
        acc[ms][ns] = __builtin_amdgcn_wmma_f32_16x16x32_bf16(
            false, a[ms], false, b, (short)0, acc[ms][ns], false, false);
    }
  }

  const int hrow0 = base[r] + m0;
  const float* b1r = b1 + (size_t)r * F_;
#pragma unroll
  for (int ns = 0; ns < 4; ++ns) {
    const int col = n0 + ns * 16 + nn;
    const float bias = b1r[col];
#pragma unroll
    for (int ms = 0; ms < 4; ++ms) {
#pragma unroll
      for (int e = 0; e < 8; ++e) {
        float v = acc[ms][ns][e] + bias;
        v = v > 0.0f ? v : 0.0f;
        H[(size_t)(hrow0 + ms * 16 + e + 8 * h) * F_ + col] = f2bf(v);
      }
    }
  }
}

// ---------------------------------------------------------------------------
// GEMM2: logits = H @ W2t + b2 -> d_out (scattered to original token rows).
// Same tiling, K = 3072.  grid = (V/256, 32, R)
// ---------------------------------------------------------------------------
__global__ __launch_bounds__(256) void gemm2_kernel(
    const unsigned short* __restrict__ H, const unsigned short* __restrict__ W2t,
    const float* __restrict__ b2, const int* __restrict__ cnt,
    const int* __restrict__ base, const int* __restrict__ perm,
    float* __restrict__ out) {
  const int r = blockIdx.z;
  const int c = cnt[r];
  const int mblk = blockIdx.y * 128;
  if (mblk >= c) return;

  const int tid = threadIdx.x, wave = tid >> 5, lane = tid & 31;
  const int h = lane >> 4, nn = lane & 15;
  const int m0 = mblk + (wave & 1) * 64;
  const int n0 = blockIdx.x * 256 + (wave >> 1) * 64;
  const int* permr = perm + r * MAXPR;
  const int hrow0 = base[r] + m0;

  const unsigned short* aptr[4];
#pragma unroll
  for (int ms = 0; ms < 4; ++ms)
    aptr[ms] = H + (size_t)(hrow0 + ms * 16 + nn) * F_;
  const unsigned short* bptr[4];
#pragma unroll
  for (int ns = 0; ns < 4; ++ns)
    bptr[ns] = W2t + ((size_t)r * V_ + (n0 + ns * 16 + nn)) * F_;

  const v8f zero = {};
  v8f acc[4][4];
#pragma unroll
  for (int ms = 0; ms < 4; ++ms)
#pragma unroll
    for (int ns = 0; ns < 4; ++ns) acc[ms][ns] = zero;

  for (int k0 = 0; k0 < F_; k0 += 32) {
    v16bf a[4];
#pragma unroll
    for (int ms = 0; ms < 4; ++ms) a[ms] = load_a(aptr[ms], k0, h);
#pragma unroll
    for (int ns = 0; ns < 4; ++ns) {
      v16bf b = load_b(bptr[ns], k0, h);
#pragma unroll
      for (int ms = 0; ms < 4; ++ms)
        acc[ms][ns] = __builtin_amdgcn_wmma_f32_16x16x32_bf16(
            false, a[ms], false, b, (short)0, acc[ms][ns], false, false);
    }
  }

  const float* b2r = b2 + (size_t)r * V_;
#pragma unroll
  for (int ns = 0; ns < 4; ++ns) {
    const int col = n0 + ns * 16 + nn;
    const float bias = b2r[col];
#pragma unroll
    for (int ms = 0; ms < 4; ++ms) {
#pragma unroll
      for (int e = 0; e < 8; ++e) {
        int row = m0 + ms * 16 + e + 8 * h;
        if (row < c) {
          int tok = permr[row];
          out[(size_t)tok * V_ + col] = acc[ms][ns][e] + bias;
        }
      }
    }
  }
}

// ---------------------------------------------------------------------------
// In-place softmax over V per token.
// ---------------------------------------------------------------------------
__global__ __launch_bounds__(256) void softmax_kernel(float* __restrict__ out) {
  __shared__ float red[256];
  float* row = out + (size_t)blockIdx.x * V_;
  const int tid = threadIdx.x;

  float m = -3.402823466e38f;
  for (int i = tid; i < V_; i += 256) m = fmaxf(m, row[i]);
  red[tid] = m;
  __syncthreads();
  for (int s = 128; s > 0; s >>= 1) {
    if (tid < s) red[tid] = fmaxf(red[tid], red[tid + s]);
    __syncthreads();
  }
  const float M = red[0];
  __syncthreads();

  float sum = 0.0f;
  for (int i = tid; i < V_; i += 256) {
    float e = __expf(row[i] - M);
    row[i] = e;
    sum += e;
  }
  red[tid] = sum;
  __syncthreads();
  for (int s = 128; s > 0; s >>= 1) {
    if (tid < s) red[tid] += red[tid + s];
    __syncthreads();
  }
  const float inv = 1.0f / red[0];
  for (int i = tid; i < V_; i += 256) row[i] *= inv;
}

// ---------------------------------------------------------------------------
extern "C" void kernel_launch(void* const* d_in, const int* in_sizes, int n_in,
                              void* d_out, int out_size, void* d_ws, size_t ws_size,
                              hipStream_t stream) {
  (void)in_sizes; (void)n_in; (void)out_size; (void)ws_size;
  const float* X   = (const float*)d_in[0];
  const int*   rid = (const int*)d_in[1];
  const float* W1  = (const float*)d_in[2];
  const float* b1  = (const float*)d_in[3];
  const float* W2  = (const float*)d_in[4];
  const float* b2  = (const float*)d_in[5];
  float*       out = (float*)d_out;

  // ---- workspace layout (bytes, 256-aligned regions) ----
  char* ws = (char*)d_ws;
  int*            cnt  = (int*)(ws + 0);                      // 8 ints
  int*            base = (int*)(ws + 64);                     // 8 ints
  int*            perm = (int*)(ws + 256);                    // 8*4096 ints
  unsigned short* Xb   = (unsigned short*)(ws + 131584);      // 4096*768
  unsigned short* W1t  = (unsigned short*)(ws + 6423296);     // 8*3072*768
  unsigned short* W2t  = (unsigned short*)(ws + 44172032);    // 8*2048*3072
  unsigned short* H    = (unsigned short*)(ws + 144835328);   // 5120*3072

  hipMemsetAsync(cnt, 0, R_ * sizeof(int), stream);

  // Convert / transpose weights & activations to bf16.
  convert_kernel<<<dim3((NTOK * D_ + 255) / 256), dim3(256), 0, stream>>>(
      X, Xb, NTOK * D_);
  transpose_convert_kernel<<<dim3(F_ / 32, D_ / 32, R_), dim3(32, 8), 0, stream>>>(
      W1, W1t, D_, F_);
  transpose_convert_kernel<<<dim3(V_ / 32, F_ / 32, R_), dim3(32, 8), 0, stream>>>(
      W2, W2t, F_, V_);

  // Route binning + padded row bases.
  route_bin_kernel<<<dim3((NTOK + 255) / 256), dim3(256), 0, stream>>>(rid, cnt, perm);
  route_base_kernel<<<dim3(1), dim3(64), 0, stream>>>(cnt, base);

  // Routed MLP.
  gemm1_kernel<<<dim3(F_ / 256, 32, R_), dim3(256), 0, stream>>>(
      Xb, W1t, b1, cnt, base, perm, H);
  gemm2_kernel<<<dim3(V_ / 256, 32, R_), dim3(256), 0, stream>>>(
      H, W2t, b2, cnt, base, perm, out);

  softmax_kernel<<<dim3(NTOK), dim3(256), 0, stream>>>(out);
}
